// SubQAttention_26233660244180
// MI455X (gfx1250) — matrix operand order repaired and checked by reading, not compiled, not run on hardware
//
#include <hip/hip_runtime.h>
#include <hip/hip_bf16.h>

// ---------------------------------------------------------------------------
// Sub-query sparse attention for MI455X (gfx1250), wave32 + WMMA bf16 +
// CDNA5 async global->LDS staging.
//   x:[4,4096,2048] f32, w_q/k/v/o:[2048,2048], w_router:[16,2048], top_k=512
// Pipeline: router WMMA -> bitonic top-k -> gathered QKV WMMA ->
//           masked softmax attention WMMA -> output proj WMMA + atomic scatter
// ---------------------------------------------------------------------------

#define Bc   4
#define Tc   4096
#define Dc   2048
#define Hc   16
#define HDc  128
#define Kc   512
#define SCALE 0.08838834764831845f   // 1/sqrt(128)
#define SCS  514                      // scores LDS row stride (shorts), 257 dwords -> conflict-free

typedef __attribute__((ext_vector_type(16))) __bf16 v16bf;
typedef __attribute__((ext_vector_type(8)))  float  v8f;

union FragU { v16bf v; unsigned int u[8]; };

__device__ __forceinline__ unsigned short f2bf(float x) {
    return __builtin_bit_cast(unsigned short, __float2bfloat16(x));
}
__device__ __forceinline__ float bf2f(unsigned short h) {
    union { unsigned int u; float f; } v; v.u = ((unsigned int)h) << 16;
    return v.f;
}

// ---- CDNA5 async global->LDS copy (ASYNCcnt-tracked, no VGPR round trip) ----
// Per CDNA5 ISA: each lane copies one dword from its global address to its LDS address.
// Generic LDS pointers carry the wave-relative LDS byte address in bits[31:0].
__device__ __forceinline__ void async_load_b32(void* lds_ptr, const void* gptr) {
    unsigned int lds_off = (unsigned int)(unsigned long long)lds_ptr;
    asm volatile("global_load_async_to_lds_b32 %0, %1, off"
                 :: "v"(lds_off), "v"(gptr) : "memory");
}
__device__ __forceinline__ void wait_async() {
    asm volatile("s_wait_asynccnt 0x0" ::: "memory");
}

// A fragment (16x32 bf16, MxK): a_lds row-major [m][k], row stride in shorts.
// ISA layout: lanes 0-15 m=0..15 {K0..7,K16..23}, lanes 16-31 {K8..15,K24..31}.
__device__ __forceinline__ v16bf load_frag_a(const unsigned short* a_lds, int stride,
                                             int m0, int kbase, int lane) {
    int m = m0 + (lane & 15);
    int g = (lane >> 4) & 1;
    const unsigned short* row = a_lds + m * stride + kbase;
    FragU f;
#pragma unroll
    for (int j = 0; j < 8; ++j) {
        int k0 = ((j < 4) ? (2 * j) : (16 + 2 * (j - 4))) + 8 * g;
        f.u[j] = *(const unsigned int*)(row + k0);
    }
    return f.v;
}

// B fragment (32x16 bf16, KxN) from b_lds[n][k] (N-major = weight-row-major).
// Lanes 0-15 hold {K0..7,K16..23} (packed pairs per VGPR), lanes 16-31 {K8..15,K24..31}.
__device__ __forceinline__ v16bf load_frag_b(const unsigned short* b_lds, int stride,
                                             int n0, int kbase, int lane) {
    int n = n0 + (lane & 15);
    int g = (lane >> 4) & 1;
    const unsigned short* row = b_lds + n * stride + kbase + 8 * g;
    FragU f;
#pragma unroll
    for (int j = 0; j < 8; ++j) f.u[j] = *(const unsigned int*)(row + ((j < 4) ? (2 * j) : (16 + 2 * (j - 4))));
    return f.v;
}

__device__ __forceinline__ v8f wmma_bf16(v16bf a, v16bf b, v8f c) {
    return __builtin_amdgcn_wmma_f32_16x16x32_bf16(false, a, false, b, (short)0, c,
                                                   false, false);
}

// ---------------------------------------------------------------------------
// 1) Router: scores[b,h,t] = x[b,t,:].w_router[h,:]   ([16384,2048]x[2048,16])
// ---------------------------------------------------------------------------
__global__ void router_kernel(const float* __restrict__ x,
                              const float* __restrict__ wr,
                              float* __restrict__ scores) {
    __shared__ unsigned short a_lds[128 * 34];
    __shared__ unsigned short b_lds[16 * 34];
    int tid = threadIdx.x, lane = tid & 31, wv = tid >> 5;
    int mbase = blockIdx.x * 128;
    v8f acc = {};
    for (int kc = 0; kc < Dc; kc += 32) {
        __syncthreads();
        {   // A: 128 rows x 32 cols of x (f32 -> bf16)
            int r = tid >> 1, c0 = (tid & 1) * 16;
            const float* src = x + (size_t)(mbase + r) * Dc + kc + c0;
            if (kc + 32 < Dc) __builtin_prefetch(src + 32, 0, 0);
            unsigned short* dst = a_lds + r * 34 + c0;
#pragma unroll
            for (int i = 0; i < 16; ++i) dst[i] = f2bf(src[i]);
        }
        if (tid < 32) {  // B: 16 router rows x 32 cols (wave 0 only: uniform)
            int r = tid >> 1, c0 = (tid & 1) * 16;
            const float* src = wr + (size_t)r * Dc + kc + c0;
            unsigned short* dst = b_lds + r * 34 + c0;
#pragma unroll
            for (int i = 0; i < 16; ++i) dst[i] = f2bf(src[i]);
        }
        __syncthreads();
        v16bf a = load_frag_a(a_lds, 34, wv * 16, 0, lane);
        v16bf b = load_frag_b(b_lds, 34, 0, 0, lane);
        acc = wmma_bf16(a, b, acc);
    }
    int n = lane & 15, g = lane >> 4;
#pragma unroll
    for (int j = 0; j < 8; ++j) {
        int m = mbase + wv * 16 + j + 8 * g;       // flat row in [0, B*T)
        int b_ = m / Tc, t = m % Tc;
        scores[((size_t)(b_ * Hc + n)) * Tc + t] = acc[j];
    }
}

// ---------------------------------------------------------------------------
// 2) Top-k: per (b,h) descending bitonic sort of 4096 (score,index) in LDS.
// ---------------------------------------------------------------------------
__global__ void topk_kernel(const float* __restrict__ scores, int* __restrict__ idx) {
    __shared__ float sv[Tc];
    __shared__ int   si[Tc];
    int bh = blockIdx.x;
    const float* s = scores + (size_t)bh * Tc;
    for (int i = threadIdx.x; i < Tc; i += 256) { sv[i] = s[i]; si[i] = i; }
    __syncthreads();
    for (int k = 2; k <= Tc; k <<= 1) {
        for (int j = k >> 1; j > 0; j >>= 1) {
            for (int i = threadIdx.x; i < Tc; i += 256) {
                int ixj = i ^ j;
                if (ixj > i) {
                    float vi = sv[i], vj = sv[ixj];
                    bool desc = ((i & k) == 0);
                    if (desc ? (vi < vj) : (vi > vj)) {
                        sv[i] = vj; sv[ixj] = vi;
                        int t = si[i]; si[i] = si[ixj]; si[ixj] = t;
                    }
                }
            }
            __syncthreads();
        }
    }
    for (int i = threadIdx.x; i < Kc; i += 256) idx[(size_t)bh * Kc + i] = si[i];
}

// ---------------------------------------------------------------------------
// 3) Gathered QKV: per (b,h,mat,mb): [64,2048]x[2048,128] into bf16 scratch.
//    grid = B*H*3*8, 8 waves: wave w -> mtile=w&3, ntiles (w>>2)*4 .. +3.
// ---------------------------------------------------------------------------
__global__ void qkv_kernel(const float* __restrict__ x,
                           const float* __restrict__ wq,
                           const float* __restrict__ wk,
                           const float* __restrict__ wvp,
                           const int* __restrict__ idx,
                           unsigned short* __restrict__ q_s,
                           unsigned short* __restrict__ k_s,
                           unsigned short* __restrict__ v_s) {
    __shared__ unsigned short a_lds[64 * 34];
    __shared__ unsigned short b_lds[128 * 34];
    __shared__ int rowtok[64];
    int blk = blockIdx.x;
    int mb = blk & 7; blk >>= 3;
    int mat = blk % 3; blk /= 3;
    int h = blk % Hc, b = blk / Hc;
    const float* w = (mat == 0) ? wq : (mat == 1) ? wk : wvp;
    unsigned short* out = (mat == 0) ? q_s : (mat == 1) ? k_s : v_s;
    int tid = threadIdx.x, lane = tid & 31, wv = tid >> 5;
    if (tid < 64) rowtok[tid] = idx[((size_t)(b * Hc + h)) * Kc + mb * 64 + tid];
    int mtile = wv & 3, nb0 = (wv >> 2) * 4;
    v8f acc[4] = {};
    for (int kc = 0; kc < Dc; kc += 32) {
        __syncthreads();
        {   // A: 64 gathered x-rows x 32 (f32 -> bf16)
            int r = tid >> 2, c0 = (tid & 3) * 8;
            const float* src = x + ((size_t)b * Tc + rowtok[r]) * Dc + kc + c0;
            if (kc + 32 < Dc) __builtin_prefetch(src + 32, 0, 0);  // global_prefetch_b8
            unsigned short* dst = a_lds + r * 34 + c0;
#pragma unroll
            for (int i = 0; i < 8; ++i) dst[i] = f2bf(src[i]);
        }
        {   // B: weight rows h*128+n, direct row copy (f32 -> bf16)
            int r = tid >> 1, c0 = (tid & 1) * 16;
            const float* src = w + ((size_t)(h * HDc + r)) * Dc + kc + c0;
            if (kc + 32 < Dc) __builtin_prefetch(src + 32, 0, 0);
            unsigned short* dst = b_lds + r * 34 + c0;
#pragma unroll
            for (int i = 0; i < 16; ++i) dst[i] = f2bf(src[i]);
        }
        __syncthreads();
        v16bf a = load_frag_a(a_lds, 34, mtile * 16, 0, lane);
#pragma unroll
        for (int q = 0; q < 4; ++q) {
            v16bf bf = load_frag_b(b_lds, 34, (nb0 + q) * 16, 0, lane);
            acc[q] = wmma_bf16(a, bf, acc[q]);
        }
    }
    int n = lane & 15, g = lane >> 4;
#pragma unroll
    for (int q = 0; q < 4; ++q)
#pragma unroll
        for (int j = 0; j < 8; ++j) {
            int mrow = mb * 64 + mtile * 16 + j + 8 * g;
            int col = (nb0 + q) * 16 + n;
            out[(((size_t)(b * Hc + h)) * Kc + mrow) * HDc + col] = f2bf(acc[q][j]);
        }
}

// ---------------------------------------------------------------------------
// 4) Attention per (b,h, 64-row block): QK^T (masked by original positions),
//    softmax (f32 stats, bf16 P), P.V -> out_s bf16.  grid = B*H*8.
//    bf16 tiles are staged via CDNA5 async global->LDS loads.
// ---------------------------------------------------------------------------
__global__ void attn_kernel(const unsigned short* __restrict__ q_s,
                            const unsigned short* __restrict__ k_s,
                            const unsigned short* __restrict__ v_s,
                            const int* __restrict__ idx,
                            unsigned short* __restrict__ out_s) {
    __shared__ unsigned short q_lds[64 * 130];
    __shared__ unsigned short kv_lds[128 * 130];
    __shared__ unsigned short sc[64 * SCS];     // bf16 scores -> probabilities
    __shared__ int posq[64];
    __shared__ int posk[Kc];
    int blk = blockIdx.x;
    int mb = blk & 7, bh = blk >> 3;
    int tid = threadIdx.x, lane = tid & 31, wv = tid >> 5;
    int mtile = wv & 3, nb0 = (wv >> 2) * 4;
    const size_t baseKV = (size_t)bh * Kc * HDc;

    if (tid < 64) posq[tid] = idx[(size_t)bh * Kc + mb * 64 + tid];
    for (int i = tid; i < Kc; i += 256) posk[i] = idx[(size_t)bh * Kc + i];
    {   // q tile 64x128 bf16: async u32 copies straight into LDS
        const unsigned int* src = (const unsigned int*)(q_s + baseKV + (size_t)mb * 64 * HDc);
        for (int i = tid; i < 64 * 64; i += 256) {
            int r = i >> 6, c = i & 63;
            async_load_b32(&q_lds[r * 130 + 2 * c], src + r * 64 + c);
        }
        wait_async();
    }
    // ---- scores ----
    for (int nch = 0; nch < 4; ++nch) {
        __syncthreads();
        const unsigned int* ks = (const unsigned int*)(k_s + baseKV + (size_t)nch * 128 * HDc);
        for (int i = tid; i < 128 * 64; i += 256) {
            int r = i >> 6, c = i & 63;
            async_load_b32(&kv_lds[r * 130 + 2 * c], ks + r * 64 + c);
        }
        wait_async();
        __syncthreads();
        v8f acc[4] = {};
#pragma unroll
        for (int kc = 0; kc < 128; kc += 32) {
            v16bf a = load_frag_a(q_lds, 130, mtile * 16, kc, lane);
#pragma unroll
            for (int q = 0; q < 4; ++q) {
                v16bf bf = load_frag_b(kv_lds, 130, (nb0 + q) * 16, kc, lane);
                acc[q] = wmma_bf16(a, bf, acc[q]);
            }
        }
        int n = lane & 15, g = lane >> 4;
#pragma unroll
        for (int q = 0; q < 4; ++q)
#pragma unroll
            for (int j = 0; j < 8; ++j) {
                int ml = mtile * 16 + j + 8 * g;
                int l = nch * 128 + (nb0 + q) * 16 + n;
                float vsc = acc[q][j] * SCALE;
                if (posk[l] > posq[ml]) vsc = -__builtin_inff();
                sc[ml * SCS + l] = f2bf(vsc);
            }
    }
    __syncthreads();
    // ---- softmax: 4 lanes per row, f32 stats ----
    {
        int r = tid >> 2, sub = tid & 3;
        float mx = -3.4e38f;
        for (int c = sub; c < Kc; c += 4) mx = fmaxf(mx, bf2f(sc[r * SCS + c]));
        mx = fmaxf(mx, __shfl_xor(mx, 1, 32));
        mx = fmaxf(mx, __shfl_xor(mx, 2, 32));
        float sum = 0.f;
        for (int c = sub; c < Kc; c += 4) sum += __expf(bf2f(sc[r * SCS + c]) - mx);
        sum += __shfl_xor(sum, 1, 32);
        sum += __shfl_xor(sum, 2, 32);
        float inv = 1.0f / sum;
        for (int c = sub; c < Kc; c += 4)
            sc[r * SCS + c] = f2bf(__expf(bf2f(sc[r * SCS + c]) - mx) * inv);
    }
    __syncthreads();
    // ---- P @ V  ([64,512]x[512,128]) ----
    v8f oacc[4] = {};
    for (int nch = 0; nch < 4; ++nch) {
        __syncthreads();
        {   // stage V chunk transposed: kv_lds[d][l_local] (regular ds writes)
            int r = tid >> 1, dbase = (tid & 1) * 64;
            const unsigned short* vs = v_s + baseKV + (size_t)(nch * 128 + r) * HDc + dbase;
#pragma unroll 8
            for (int i = 0; i < 64; ++i) kv_lds[(dbase + i) * 130 + r] = vs[i];
        }
        __syncthreads();
#pragma unroll
        for (int kc = 0; kc < 128; kc += 32) {
            v16bf a = load_frag_a(sc, SCS, mtile * 16, nch * 128 + kc, lane);
#pragma unroll
            for (int q = 0; q < 4; ++q) {
                v16bf bf = load_frag_b(kv_lds, 130, (nb0 + q) * 16, kc, lane);
                oacc[q] = wmma_bf16(a, bf, oacc[q]);
            }
        }
    }
    int n = lane & 15, g = lane >> 4;
#pragma unroll
    for (int q = 0; q < 4; ++q)
#pragma unroll
        for (int j = 0; j < 8; ++j) {
            int mrow = mb * 64 + mtile * 16 + j + 8 * g;
            int col = (nb0 + q) * 16 + n;
            out_s[((size_t)bh * Kc + mrow) * HDc + col] = f2bf(oacc[q][j]);
        }
}

// ---------------------------------------------------------------------------
// 5) Zero output, then scatter output projection with atomic adds.
// ---------------------------------------------------------------------------
__global__ void zero_kernel(float* __restrict__ out) {
    size_t i = ((size_t)blockIdx.x * 256 + threadIdx.x) * 4;
    *(float4*)(out + i) = make_float4(0.f, 0.f, 0.f, 0.f);
}

__global__ void oproj_kernel(const unsigned short* __restrict__ out_s,
                             const float* __restrict__ wo,
                             const int* __restrict__ idx,
                             float* __restrict__ out) {
    __shared__ unsigned short a_lds[64 * 130];
    __shared__ unsigned short b_lds[128 * 130];
    __shared__ int rowtok[64];
    int blk = blockIdx.x;
    int nb = blk & 15; blk >>= 4;
    int mb = blk & 7;  blk >>= 3;
    int bh = blk, h = bh & 15, b = bh >> 4;
    int tid = threadIdx.x, lane = tid & 31, wv = tid >> 5;
    int mtile = wv & 3, nb0 = (wv >> 2) * 4;
    if (tid < 64) rowtok[tid] = idx[(size_t)bh * Kc + mb * 64 + tid];
    {   // A: out_s tile, async bf16 u32 copies straight into LDS
        const unsigned int* src =
            (const unsigned int*)(out_s + ((size_t)bh * Kc + mb * 64) * HDc);
        for (int i = tid; i < 64 * 64; i += 256) {
            int r = i >> 6, c = i & 63;
            async_load_b32(&a_lds[r * 130 + 2 * c], src + r * 64 + c);
        }
    }
    {   // B: w_o rows nb*128+n, cols h*128..+127 (direct row copy, f32->bf16)
        int r = tid >> 1, c0 = (tid & 1) * 64;
        const float* src = wo + (size_t)(nb * 128 + r) * Dc + h * HDc + c0;
        unsigned short* dst = b_lds + r * 130 + c0;
#pragma unroll 8
        for (int i = 0; i < 64; ++i) dst[i] = f2bf(src[i]);
    }
    wait_async();
    __syncthreads();
    v8f acc[4] = {};
#pragma unroll
    for (int kc = 0; kc < 128; kc += 32) {
        v16bf a = load_frag_a(a_lds, 130, mtile * 16, kc, lane);
#pragma unroll
        for (int q = 0; q < 4; ++q) {
            v16bf bf = load_frag_b(b_lds, 130, (nb0 + q) * 16, kc, lane);
            acc[q] = wmma_bf16(a, bf, acc[q]);
        }
    }
    int n = lane & 15, g = lane >> 4;
#pragma unroll
    for (int q = 0; q < 4; ++q)
#pragma unroll
        for (int j = 0; j < 8; ++j) {
            int token = rowtok[mtile * 16 + j + 8 * g];
            int col = nb * 128 + (nb0 + q) * 16 + n;
            atomicAdd(out + ((size_t)b * Tc + token) * Dc + col, acc[q][j]);
        }
}

// ---------------------------------------------------------------------------
extern "C" void kernel_launch(void* const* d_in, const int* in_sizes, int n_in,
                              void* d_out, int out_size, void* d_ws, size_t ws_size,
                              hipStream_t stream) {
    (void)in_sizes; (void)n_in; (void)out_size; (void)ws_size;
    const float* x  = (const float*)d_in[0];
    const float* wq = (const float*)d_in[1];
    const float* wk = (const float*)d_in[2];
    const float* wv = (const float*)d_in[3];
    const float* wo = (const float*)d_in[4];
    const float* wr = (const float*)d_in[5];
    float* out = (float*)d_out;

    // workspace layout (~33.2 MB total)
    char* ws = (char*)d_ws;
    float* scores = (float*)ws;                                  // 1 MiB
    int* idxbuf = (int*)(ws + (1 << 20));                        // 128 KiB
    unsigned short* q_s = (unsigned short*)(ws + (1 << 20) + (128 << 10));
    unsigned short* k_s = q_s + (size_t)Bc * Hc * Kc * HDc;      // 8 MiB each
    unsigned short* v_s = k_s + (size_t)Bc * Hc * Kc * HDc;
    unsigned short* o_s = v_s + (size_t)Bc * Hc * Kc * HDc;

    router_kernel<<<(Bc * Tc) / 128, 256, 0, stream>>>(x, wr, scores);
    topk_kernel<<<Bc * Hc, 256, 0, stream>>>(scores, idxbuf);
    qkv_kernel<<<Bc * Hc * 3 * 8, 256, 0, stream>>>(x, wq, wk, wv, idxbuf, q_s, k_s, v_s);
    attn_kernel<<<Bc * Hc * 8, 256, 0, stream>>>(q_s, k_s, v_s, idxbuf, o_s);
    zero_kernel<<<(Bc * Tc * (Dc / 4)) / 256, 256, 0, stream>>>(out);
    oproj_kernel<<<Bc * Hc * 8 * 16, 256, 0, stream>>>(o_s, wo, idxbuf, out);
}
